// GNNBlock_45414984188101
// MI455X (gfx1250) — matrix operand order repaired and checked by reading, not compiled, not run on hardware
//
#include <hip/hip_runtime.h>
#include <hip/hip_bf16.h>
#include <math.h>

typedef __bf16 bf16_t;
typedef __attribute__((ext_vector_type(16))) __bf16 v16bf;
typedef __attribute__((ext_vector_type(8)))  __bf16 v8bf;
typedef __attribute__((ext_vector_type(8)))  float  v8f;
typedef __attribute__((ext_vector_type(4)))  float  v4f;

#define WAVES 8

__device__ __forceinline__ float gelu_f(float x) {
    const float c = 0.7978845608028654f;      // sqrt(2/pi)
    float x3 = x * x * x;
    return 0.5f * x * (1.0f + tanhf(c * (x + 0.044715f * x3)));
}

__global__ void zero_kernel(float* __restrict__ p, int n) {
    int i = blockIdx.x * blockDim.x + threadIdx.x;
    if (i < n) p[i] = 0.0f;
}

// ---------------------------------------------------------------------------
// Edge MLP: per 16-edge tile  A(16x192) -> gelu(A*We1+be1) -> (*We2+be2)
// then atomic scatter-add (scaled by 1/sqrt(16)) into m_i[N,64].
// ---------------------------------------------------------------------------
__global__ __launch_bounds__(256) void edge_mlp_kernel(
    const float* __restrict__ nf, const float* __restrict__ ef,
    const int*  __restrict__ senders, const int* __restrict__ receivers,
    const float* __restrict__ We1, const float* __restrict__ be1,
    const float* __restrict__ We2, const float* __restrict__ be2,
    float* __restrict__ mi, int n_tiles)
{
    __shared__ bf16_t WT1[192 * 192];          // We1 transposed: [n][k]
    __shared__ bf16_t WT2[64 * 192];           // We2 transposed: [n][k]
    __shared__ bf16_t hst[WAVES][16 * 192];    // per-wave h staging [m][k]
    __shared__ int    ridx[WAVES][16];

    const int tid = threadIdx.x;

    // Cooperative transposed bf16 weight load (reads coalesced).
    for (int i = tid; i < 192 * 192; i += 256) {
        int k = i / 192, n = i % 192;
        WT1[n * 192 + k] = (bf16_t)We1[i];
    }
    for (int i = tid; i < 64 * 192; i += 256) {
        int k = i / 64, n = i % 64;
        WT2[n * 192 + k] = (bf16_t)We2[i];
    }
    __syncthreads();

    const int lane = tid & 31;
    const int wave = tid >> 5;
    const int row  = lane & 15;     // M row (A) / N col (B,C,D)
    const int hi   = lane >> 4;     // lane half selects K sub-chunk

    for (int tile = blockIdx.x * WAVES + wave; tile < n_tiles;
         tile += gridDim.x * WAVES) {
        const int e  = tile * 16 + row;
        const int si = senders[e];
        const int ri = receivers[e];
        if (hi == 0) ridx[wave][row] = ri;

        const float* src0 = nf + (size_t)si * 64;
        const float* src1 = nf + (size_t)ri * 64;
        const float* src2 = ef + (size_t)e  * 64;

        // A fragments: 16x192 bf16 across 6 k-steps (ISA 16-bit A layout).
        v16bf afrag[6];
        #pragma unroll
        for (int s = 0; s < 6; ++s) {
            int base = s * 32;
            const float* sp = (base < 64) ? src0 : (base < 128) ? src1 : src2;
            int off = base & 63;
            int ka  = off + hi * 8;
            int kb  = ka + 16;
            v4f f0 = *(const v4f*)(sp + ka);
            v4f f1 = *(const v4f*)(sp + ka + 4);
            v4f f2 = *(const v4f*)(sp + kb);
            v4f f3 = *(const v4f*)(sp + kb + 4);
            v16bf a;
            #pragma unroll
            for (int j = 0; j < 4; ++j) {
                a[j]      = (bf16_t)f0[j];
                a[4 + j]  = (bf16_t)f1[j];
                a[8 + j]  = (bf16_t)f2[j];
                a[12 + j] = (bf16_t)f3[j];
            }
            afrag[s] = a;
        }

        bf16_t* hrow = &hst[wave][0];

        // GEMM1: h = gelu(A * We1 + be1), 12 column blocks of 16.
        for (int nb = 0; nb < 12; ++nb) {
            v8f acc = {};
            const bf16_t* wp = &WT1[(nb * 16 + row) * 192];
            #pragma unroll
            for (int s = 0; s < 6; ++s) {
                v16bf b = __builtin_shufflevector(*(const v8bf*)(wp + s * 32 + hi * 8), *(const v8bf*)(wp + s * 32 + hi * 8 + 16), 0, 1, 2, 3, 4, 5, 6, 7, 8, 9, 10, 11, 12, 13, 14, 15);
                acc = __builtin_amdgcn_wmma_f32_16x16x32_bf16(
                        false, afrag[s], false, b, (short)0, acc, false, false);
            }
            float bias = be1[nb * 16 + row];
            #pragma unroll
            for (int v = 0; v < 8; ++v) {
                float g = gelu_f(acc[v] + bias);
                hrow[(v + hi * 8) * 192 + nb * 16 + row] = (bf16_t)g;
            }
        }
        asm volatile("s_wait_dscnt 0" ::: "memory");  // DS in-order within wave

        // Re-load h in A-fragment layout from staging.
        v16bf a2[6];
        #pragma unroll
        for (int s = 0; s < 6; ++s) {
            int ka = s * 32 + hi * 8;
            v8bf lo = *(const v8bf*)(hrow + row * 192 + ka);
            v8bf hb = *(const v8bf*)(hrow + row * 192 + ka + 16);
            v16bf a;
            #pragma unroll
            for (int j = 0; j < 8; ++j) { a[j] = lo[j]; a[8 + j] = hb[j]; }
            a2[s] = a;
        }

        // GEMM2: m = h * We2 + be2; scatter (scaled 1/4) to receivers.
        for (int nb = 0; nb < 4; ++nb) {
            v8f acc = {};
            const bf16_t* wp = &WT2[(nb * 16 + row) * 192];
            #pragma unroll
            for (int s = 0; s < 6; ++s) {
                v16bf b = __builtin_shufflevector(*(const v8bf*)(wp + s * 32 + hi * 8), *(const v8bf*)(wp + s * 32 + hi * 8 + 16), 0, 1, 2, 3, 4, 5, 6, 7, 8, 9, 10, 11, 12, 13, 14, 15);
                acc = __builtin_amdgcn_wmma_f32_16x16x32_bf16(
                        false, a2[s], false, b, (short)0, acc, false, false);
            }
            float bias = be2[nb * 16 + row];
            #pragma unroll
            for (int v = 0; v < 8; ++v) {
                int m_row = v + hi * 8;
                int rv    = ridx[wave][m_row];
                atomicAdd(&mi[(size_t)rv * 64 + nb * 16 + row],
                          (acc[v] + bias) * 0.25f);
            }
        }
        asm volatile("s_wait_dscnt 0" ::: "memory");  // hst reuse next tile
    }
}

// ---------------------------------------------------------------------------
// Node MLP: per 16-node tile  A=[m_i | x] (16x128) -> gelu(A*Wn1+bn1) -> *Wn2+bn2
// ---------------------------------------------------------------------------
__global__ __launch_bounds__(256) void node_mlp_kernel(
    const float* __restrict__ mi, const float* __restrict__ nf,
    const float* __restrict__ Wn1, const float* __restrict__ bn1,
    const float* __restrict__ Wn2, const float* __restrict__ bn2,
    float* __restrict__ out, int n_tiles)
{
    __shared__ bf16_t WT1[128 * 128];
    __shared__ bf16_t WT2[64 * 128];
    __shared__ bf16_t hst[WAVES][16 * 128];

    const int tid = threadIdx.x;
    for (int i = tid; i < 128 * 128; i += 256) {
        int k = i / 128, n = i % 128;
        WT1[n * 128 + k] = (bf16_t)Wn1[i];
    }
    for (int i = tid; i < 64 * 128; i += 256) {
        int k = i / 64, n = i % 64;
        WT2[n * 128 + k] = (bf16_t)Wn2[i];
    }
    __syncthreads();

    const int lane = tid & 31;
    const int wave = tid >> 5;
    const int row  = lane & 15;
    const int hi   = lane >> 4;

    for (int tile = blockIdx.x * WAVES + wave; tile < n_tiles;
         tile += gridDim.x * WAVES) {
        const int node = tile * 16 + row;
        const float* src0 = mi + (size_t)node * 64;
        const float* src1 = nf + (size_t)node * 64;

        v16bf afrag[4];
        #pragma unroll
        for (int s = 0; s < 4; ++s) {
            int base = s * 32;
            const float* sp = (base < 64) ? src0 : src1;
            int off = base & 63;
            int ka  = off + hi * 8;
            int kb  = ka + 16;
            v4f f0 = *(const v4f*)(sp + ka);
            v4f f1 = *(const v4f*)(sp + ka + 4);
            v4f f2 = *(const v4f*)(sp + kb);
            v4f f3 = *(const v4f*)(sp + kb + 4);
            v16bf a;
            #pragma unroll
            for (int j = 0; j < 4; ++j) {
                a[j]      = (bf16_t)f0[j];
                a[4 + j]  = (bf16_t)f1[j];
                a[8 + j]  = (bf16_t)f2[j];
                a[12 + j] = (bf16_t)f3[j];
            }
            afrag[s] = a;
        }

        bf16_t* hrow = &hst[wave][0];

        for (int nb = 0; nb < 8; ++nb) {
            v8f acc = {};
            const bf16_t* wp = &WT1[(nb * 16 + row) * 128];
            #pragma unroll
            for (int s = 0; s < 4; ++s) {
                v16bf b = __builtin_shufflevector(*(const v8bf*)(wp + s * 32 + hi * 8), *(const v8bf*)(wp + s * 32 + hi * 8 + 16), 0, 1, 2, 3, 4, 5, 6, 7, 8, 9, 10, 11, 12, 13, 14, 15);
                acc = __builtin_amdgcn_wmma_f32_16x16x32_bf16(
                        false, afrag[s], false, b, (short)0, acc, false, false);
            }
            float bias = bn1[nb * 16 + row];
            #pragma unroll
            for (int v = 0; v < 8; ++v) {
                float g = gelu_f(acc[v] + bias);
                hrow[(v + hi * 8) * 128 + nb * 16 + row] = (bf16_t)g;
            }
        }
        asm volatile("s_wait_dscnt 0" ::: "memory");

        v16bf a2[4];
        #pragma unroll
        for (int s = 0; s < 4; ++s) {
            int ka = s * 32 + hi * 8;
            v8bf lo = *(const v8bf*)(hrow + row * 128 + ka);
            v8bf hb = *(const v8bf*)(hrow + row * 128 + ka + 16);
            v16bf a;
            #pragma unroll
            for (int j = 0; j < 8; ++j) { a[j] = lo[j]; a[8 + j] = hb[j]; }
            a2[s] = a;
        }

        for (int nb = 0; nb < 4; ++nb) {
            v8f acc = {};
            const bf16_t* wp = &WT2[(nb * 16 + row) * 128];
            #pragma unroll
            for (int s = 0; s < 4; ++s) {
                v16bf b = __builtin_shufflevector(*(const v8bf*)(wp + s * 32 + hi * 8), *(const v8bf*)(wp + s * 32 + hi * 8 + 16), 0, 1, 2, 3, 4, 5, 6, 7, 8, 9, 10, 11, 12, 13, 14, 15);
                acc = __builtin_amdgcn_wmma_f32_16x16x32_bf16(
                        false, a2[s], false, b, (short)0, acc, false, false);
            }
            float bias = bn2[nb * 16 + row];
            #pragma unroll
            for (int v = 0; v < 8; ++v) {
                int m_row = v + hi * 8;
                out[(size_t)(tile * 16 + m_row) * 64 + nb * 16 + row] =
                    acc[v] + bias;
            }
        }
        asm volatile("s_wait_dscnt 0" ::: "memory");
    }
}

extern "C" void kernel_launch(void* const* d_in, const int* in_sizes, int n_in,
                              void* d_out, int out_size, void* d_ws, size_t ws_size,
                              hipStream_t stream) {
    const float* nf  = (const float*)d_in[0];   // (1,N,64)
    const float* ef  = (const float*)d_in[1];   // (1,E,64)
    const int*   snd = (const int*)  d_in[2];   // (1,E)
    const int*   rcv = (const int*)  d_in[3];   // (1,E)
    const float* We1 = (const float*)d_in[4];   // (192,192)
    const float* be1 = (const float*)d_in[5];   // (192)
    const float* We2 = (const float*)d_in[6];   // (192,64)
    const float* be2 = (const float*)d_in[7];   // (64)
    const float* Wn1 = (const float*)d_in[8];   // (128,128)
    const float* bn1 = (const float*)d_in[9];   // (128)
    const float* Wn2 = (const float*)d_in[10];  // (128,64)
    const float* bn2 = (const float*)d_in[11];  // (64)

    const int N = in_sizes[0] / 64;
    const int E = in_sizes[1] / 64;

    float* mi  = (float*)d_ws;                  // N*64 floats scratch
    float* out = (float*)d_out;

    const int mi_n = N * 64;
    zero_kernel<<<(mi_n + 255) / 256, 256, 0, stream>>>(mi, mi_n);

    const int e_tiles = E / 16;                 // E % 16 == 0 for this problem
    edge_mlp_kernel<<<320, 256, 0, stream>>>(nf, ef, snd, rcv,
                                             We1, be1, We2, be2, mi, e_tiles);

    const int n_tiles = N / 16;                 // N % 16 == 0 for this problem
    node_mlp_kernel<<<200, 256, 0, stream>>>(mi, nf, Wn1, bn1, Wn2, bn2,
                                             out, n_tiles);
}
